// CRD_74818330296985
// MI455X (gfx1250) — matrix operand order repaired and checked by reading, not compiled, not run on hardware
//
#include <hip/hip_runtime.h>
#include <hip/hip_bf16.h>

typedef __attribute__((ext_vector_type(16))) __bf16 v16bf;
typedef __attribute__((ext_vector_type(8)))  float  v8f;

#define IN_F  512
#define OUT_F 512

// ---------------- degree accumulation (f32 atomics, L2 resident) ----------------
__global__ void gcn_deg_kernel(const int* __restrict__ src, const int* __restrict__ dst,
                               float* __restrict__ degS, float* __restrict__ degD, int nE) {
    int e = blockIdx.x * blockDim.x + threadIdx.x;
    if (e < nE) {
        atomicAdd(&degS[src[e]], 1.0f);
        atomicAdd(&degD[dst[e]], 1.0f);
    }
}

// ---------------- norm = deg^{-1/2} (deg==0 -> 1) ----------------
__global__ void gcn_norm_kernel(const float* __restrict__ degS, const float* __restrict__ degD,
                                float* __restrict__ normS, float* __restrict__ normD, int nV) {
    int i = blockIdx.x * blockDim.x + threadIdx.x;
    if (i < nV) {
        float ds = degS[i];
        float dd = degD[i];
        normS[i] = rsqrtf(ds > 0.0f ? ds : 1.0f);
        normD[i] = rsqrtf(dd > 0.0f ? dd : 1.0f);
    }
}

// ---------------- edge scatter: agg[dst] += x[src] * normS[src] ----------------
// one block (128 threads) per edge; each thread moves a float4 (512 feats total)
__global__ __launch_bounds__(128)
void gcn_scatter_kernel(const float* __restrict__ x, const int* __restrict__ src,
                        const int* __restrict__ dst, const float* __restrict__ normS,
                        float* __restrict__ agg, int nE) {
    int e = blockIdx.x;
    if (e >= nE) return;
    int s = src[e];
    int d = dst[e];
    float ns = normS[s];
    int t = threadIdx.x;                       // 0..127, 4 floats each
    const float4 v = ((const float4*)(x + (size_t)s * IN_F))[t];
    float* o = agg + (size_t)d * IN_F + (size_t)t * 4;
    atomicAdd(o + 0, v.x * ns);
    atomicAdd(o + 1, v.y * ns);
    atomicAdd(o + 2, v.z * ns);
    atomicAdd(o + 3, v.w * ns);
}

// ---------------- WMMA GEMM: out = relu((agg * normD) @ W + b) ----------------
// block = 4 waves (32x4). All waves share one 16-col slab of W staged in LDS,
// pre-swizzled into the exact B-fragment layout (packed bf16 pairs, 1 dword/lane)
// so the K-loop does conflict-free ds_load_b32. Each wave owns a 16x16 C tile.
__global__ __launch_bounds__(128)
void gcn_gemm_wmma_kernel(const float* __restrict__ agg, const float* __restrict__ W,
                          const float* __restrict__ bias, const float* __restrict__ normD,
                          float* __restrict__ out, int nRows) {
    // 16 k-steps x 8 frag-VGPRs x 32 lanes x 2 bf16 = 8192 bf16 = 16 KB
    __shared__ __bf16 ldsB[16 * 8 * 32 * 2];

    const int tid   = threadIdx.y * 32 + threadIdx.x;   // 0..127
    const int tileN = blockIdx.x * 16;

    // Cooperative W slab load, converting f32 -> bf16 and swizzling into
    // B-fragment order: pair index p = kb*256 + v*32 + lane,
    // holds W[K, tileN+n], W[K+1, tileN+n] with K = kb*32 + h*16 + 2v,
    // n = lane&15, h = lane>>4  (contiguous k = 16h + s, same as A).
    for (int i = 0; i < 32; ++i) {
        int p  = i * 128 + tid;          // 0..4095
        int kb = p >> 8;
        int r  = p & 255;
        int vv = r >> 5;
        int l  = r & 31;
        int n  = l & 15;
        int h  = l >> 4;
        int K  = kb * 32 + h * 16 + vv * 2;
        ldsB[p * 2 + 0] = (__bf16)W[(size_t)K * OUT_F + tileN + n];
        ldsB[p * 2 + 1] = (__bf16)W[(size_t)(K + 1) * OUT_F + tileN + n];
    }
    __syncthreads();

    const int tileM = blockIdx.y * 64 + threadIdx.y * 16;
    if (tileM >= nRows) return;          // wave-uniform: EXEC stays all-ones for WMMA

    const int lane = threadIdx.x;
    const int n    = lane & 15;
    const int h    = lane >> 4;
    const int row  = tileM + n;          // A-fragment: m = lane&15
    const float nd = normD[row];
    const float* arow = agg + (size_t)row * IN_F;

    v8f acc = {};
    for (int kb = 0; kb < 16; ++kb) {
        if (kb + 1 < 16)                  // global_prefetch_b8 for next A chunk
            __builtin_prefetch(arow + (kb + 1) * 32, 0, 0);
        v16bf afrag, bfrag;
#pragma unroll
        for (int v = 0; v < 8; ++v) {
            // A 16x32 layout (matches B): K = h*16 + v*2
            int K = kb * 32 + (h << 4) + (v << 1);
            float2 p2 = *(const float2*)(arow + K);
            afrag[2 * v + 0] = (__bf16)(p2.x * nd);
            afrag[2 * v + 1] = (__bf16)(p2.y * nd);
            const __bf16* bp = &ldsB[(kb * 256 + v * 32 + lane) * 2];
            bfrag[2 * v + 0] = bp[0];
            bfrag[2 * v + 1] = bp[1];
        }
        acc = __builtin_amdgcn_wmma_f32_16x16x32_bf16(
                  false, afrag, false, bfrag, (short)0, acc, false, false);
    }

    // epilogue: C/D layout — VGPR r holds m = r + 8*h, col = tileN + n
    const float bn = bias[tileN + n];
#pragma unroll
    for (int r = 0; r < 8; ++r) {
        int mrow = tileM + r + 8 * h;
        float val = acc[r] + bn;
        out[(size_t)mrow * OUT_F + tileN + n] = val > 0.0f ? val : 0.0f;
    }
}

extern "C" void kernel_launch(void* const* d_in, const int* in_sizes, int n_in,
                              void* d_out, int out_size, void* d_ws, size_t ws_size,
                              hipStream_t stream) {
    const float* x    = (const float*)d_in[0];
    const int*   src  = (const int*)d_in[1];
    const int*   dst  = (const int*)d_in[2];
    const float* W    = (const float*)d_in[3];
    const float* bias = (const float*)d_in[4];
    float*       out  = (float*)d_out;

    const int nNodes = in_sizes[0] / IN_F;   // 50000
    const int nEdges = in_sizes[1];          // 150000

    // workspace layout (bytes):
    //   [0, 4N)        degS        [4N, 8N)   degD
    //   [8N, 12N)      normS       [12N, 16N) normD
    //   [16N, ...)     agg (nNodes x IN_F f32), 16-byte aligned (16N % 16 == 0)
    char* ws = (char*)d_ws;
    float* degS  = (float*)(ws);
    float* degD  = (float*)(ws + (size_t)4  * nNodes);
    float* normS = (float*)(ws + (size_t)8  * nNodes);
    float* normD = (float*)(ws + (size_t)12 * nNodes);
    float* agg   = (float*)(ws + (size_t)16 * nNodes);

    // zero accumulators every launch (ws is not re-cleared between replays)
    hipMemsetAsync(degS, 0, (size_t)8 * nNodes, stream);
    hipMemsetAsync(agg,  0, (size_t)nNodes * IN_F * sizeof(float), stream);

    gcn_deg_kernel<<<(nEdges + 255) / 256, 256, 0, stream>>>(src, dst, degS, degD, nEdges);
    gcn_norm_kernel<<<(nNodes + 255) / 256, 256, 0, stream>>>(degS, degD, normS, normD, nNodes);
    gcn_scatter_kernel<<<nEdges, 128, 0, stream>>>(x, src, dst, normS, agg, nEdges);

    dim3 gblk(32, 4);
    dim3 ggrd(OUT_F / 16, (nNodes + 63) / 64);
    gcn_gemm_wmma_kernel<<<ggrd, gblk, 0, stream>>>(agg, W, bias, normD, out, nNodes);
}
